// Head_24807731101992
// MI455X (gfx1250) — matrix operand order repaired and checked by reading, not compiled, not run on hardware
//
#include <hip/hip_runtime.h>

typedef __attribute__((ext_vector_type(16))) _Float16 v16h;
typedef __attribute__((ext_vector_type(8)))  _Float16 v8h;
typedef __attribute__((ext_vector_type(8)))  float    v8f;
typedef __attribute__((ext_vector_type(4)))  float    v4f;

#define WMMA_F16(a,b,c) __builtin_amdgcn_wmma_f32_16x16x32_f16(false,(a),false,(b),(short)0,(c),false,false)

constexpr int B_ = 8, T_ = 2048, E_ = 1024, D_ = 64;
constexpr int BT_ = B_ * T_;

// ---------------------------------------------------------------------------
// Kernel 0: convert Wq/Wk/Wv [E,D] f32 -> wt [3][D][E] f16 (transposed)
// ---------------------------------------------------------------------------
__global__ void wconvert_kernel(const float* __restrict__ Wq,
                                const float* __restrict__ Wk,
                                const float* __restrict__ Wv,
                                _Float16* __restrict__ wt) {
    int idx = blockIdx.x * blockDim.x + threadIdx.x;   // 3*D*E total
    if (idx >= 3 * D_ * E_) return;
    int m = idx / (D_ * E_);
    int r = idx % (D_ * E_);
    int d = r / E_;
    int e = r % E_;
    const float* W = (m == 0) ? Wq : ((m == 1) ? Wk : Wv);
    wt[idx] = (_Float16)W[e * D_ + d];
}

// ---------------------------------------------------------------------------
// Kernel 1: QKV projection. 128-thread blocks = 4 waves; each wave owns one
// 16-row tile. Per k=32 step the block cooperatively stages the 12 KB weight
// slice (3 mats x 64 cols x 32 k) into LDS; every wave then runs 12 WMMAs
// with B fragments from LDS. Cuts global weight traffic 4x.
// q stored pre-scaled by 1/sqrt(D); v stored transposed [B][D][T].
// ---------------------------------------------------------------------------
__global__ void qkv_proj_kernel(const float* __restrict__ x,
                                const _Float16* __restrict__ wt,   // [3][D][E]
                                _Float16* __restrict__ qh,         // [BT][D]
                                _Float16* __restrict__ kh,         // [BT][D]
                                _Float16* __restrict__ vt) {       // [B][D][T]
    __shared__ _Float16 wbuf[3 * 64 * 32];   // 12 KB: [m][d][k'] slice

    const int tid     = threadIdx.x;
    const int wid     = tid >> 5;
    const int lane    = tid & 31;
    const int half_id = lane >> 4;
    const int l16     = lane & 15;
    const int koff    = half_id * 16;
    const int r0      = (blockIdx.x * 4 + wid) * 16;
    const int row     = r0 + l16;

    v8f acc[3][4];
    #pragma unroll
    for (int m = 0; m < 3; ++m)
        #pragma unroll
        for (int n = 0; n < 4; ++n)
            #pragma unroll
            for (int i = 0; i < 8; ++i) acc[m][n][i] = 0.0f;

    const float* xrow = x + (size_t)row * E_;
    for (int e0 = 0; e0 < E_; e0 += 32) {
        __syncthreads();
        // Cooperative stage: 6144 halves = 768 x v8h chunks over 128 threads.
        #pragma unroll
        for (int c = tid; c < 768; c += 128) {
            int m  = c >> 8;          // 0..2
            int d  = (c >> 2) & 63;   // 0..63
            int ep = (c & 3) * 8;     // 0,8,16,24
            *(v8h*)&wbuf[(m * 64 + d) * 32 + ep] =
                *(const v8h*)(wt + ((size_t)m * D_ + d) * E_ + e0 + ep);
        }
        __syncthreads();

        // A fragment: row = lane%16, K = koff + j (koff = 16*half_id)
        v4f a0 = *(const v4f*)(xrow + e0 + koff);
        v4f a1 = *(const v4f*)(xrow + e0 + koff + 4);
        v4f a2 = *(const v4f*)(xrow + e0 + koff + 8);
        v4f a3 = *(const v4f*)(xrow + e0 + koff + 12);
        v16h a;
        #pragma unroll
        for (int j = 0; j < 4; ++j) {
            a[j]      = (_Float16)a0[j];
            a[4 + j]  = (_Float16)a1[j];
            a[8 + j]  = (_Float16)a2[j];
            a[12 + j] = (_Float16)a3[j];
        }
        #pragma unroll
        for (int m = 0; m < 3; ++m) {
            #pragma unroll
            for (int n = 0; n < 4; ++n) {
                int d = n * 16 + l16;  // B fragment: lane = col, K = 16*half_id + j
                v16h bb = *(const v16h*)&wbuf[(m * 64 + d) * 32 + half_id * 16];
                acc[m][n] = WMMA_F16(a, bb, acc[m][n]);
            }
        }
    }

    const float qscale = 0.125f;   // 1/sqrt(D), folded into Q
    #pragma unroll
    for (int n = 0; n < 4; ++n) {
        int d = n * 16 + l16;
        #pragma unroll
        for (int i = 0; i < 8; ++i) {
            int rr = r0 + i + 8 * half_id;            // C layout: row = i + 8*(lane/16)
            qh[(size_t)rr * D_ + d] = (_Float16)(acc[0][n][i] * qscale);
            kh[(size_t)rr * D_ + d] = (_Float16)(acc[1][n][i]);
            int b = rr >> 11, s = rr & (T_ - 1);
            vt[((size_t)b * D_ + d) * T_ + s] = (_Float16)(acc[2][n][i]);
        }
    }
}

// ---------------------------------------------------------------------------
// Kernel 2: causal flash attention. 128-thread blocks = 4 independent waves.
// Strided t-tile assignment (tt = wid*32 + blockIdx.x) balances causal work
// across blocks. Each wave: stream key chunks of 32, online softmax, P goes
// through a private 1 KB LDS slice to convert C-layout -> A-layout for PV.
// ---------------------------------------------------------------------------
__global__ void attn_kernel(const _Float16* __restrict__ qh,   // [BT][D] (pre-scaled)
                            const _Float16* __restrict__ kh,   // [BT][D]
                            const _Float16* __restrict__ vt,   // [B][D][T]
                            float* __restrict__ out) {         // [B][T][D]
    __shared__ _Float16 ldsP[4][16 * 32];   // per-wave P tile

    const int tid     = threadIdx.x;
    const int wid     = tid >> 5;
    const int lane    = tid & 31;
    const int half_id = lane >> 4;
    const int l16     = lane & 15;
    const int koff    = half_id * 16;
    const int b       = blockIdx.y;
    const int tt      = wid * 32 + blockIdx.x;   // strided for load balance
    const int t0      = tt * 16;
    _Float16* myP     = &ldsP[wid][0];

    // Q A-fragments for both k=32 halves of D=64 (resident for whole pass)
    const _Float16* qp = qh + ((size_t)b * T_ + (t0 + l16)) * D_;
    v16h qa[2];
    #pragma unroll
    for (int kk = 0; kk < 2; ++kk) {
        v8h lo = *(const v8h*)(qp + kk * 32 + koff);
        v8h hi = *(const v8h*)(qp + kk * 32 + koff + 8);
        #pragma unroll
        for (int j = 0; j < 8; ++j) { qa[kk][j] = lo[j]; qa[kk][8 + j] = hi[j]; }
    }

    v8f o[4];
    #pragma unroll
    for (int n = 0; n < 4; ++n)
        #pragma unroll
        for (int i = 0; i < 8; ++i) o[n][i] = 0.0f;
    float mrow[8], lrow[8];
    #pragma unroll
    for (int i = 0; i < 8; ++i) { mrow[i] = -3.0e38f; lrow[i] = 0.0f; }

    const int nchunks = (t0 + 16 + 31) >> 5;
    for (int c = 0; c < nchunks; ++c) {
        const int s0 = c * 32;

        // S = Q K^T for two 16-col subtiles (each: two k=32 WMMAs over D)
        v8f sfrag[2];
        #pragma unroll
        for (int sub = 0; sub < 2; ++sub) {
            int srow = s0 + sub * 16 + l16;   // B fragment: lane = col s
            const _Float16* kp = kh + ((size_t)b * T_ + srow) * D_ + half_id * 16;
            v16h kb0 = *(const v16h*)(kp);
            v16h kb1 = *(const v16h*)(kp + 32);
            v8f sacc;
            #pragma unroll
            for (int i = 0; i < 8; ++i) sacc[i] = 0.0f;
            sacc = WMMA_F16(qa[0], kb0, sacc);
            sacc = WMMA_F16(qa[1], kb1, sacc);
            #pragma unroll
            for (int i = 0; i < 8; ++i) {                 // causal mask
                int t = t0 + i + 8 * half_id;
                int s = s0 + sub * 16 + l16;
                if (s > t) sacc[i] = -1.0e30f;
            }
            sfrag[sub] = sacc;
        }

        // Online softmax (row stats live in the lanes owning that row)
        float p0[8], p1[8];
        #pragma unroll
        for (int i = 0; i < 8; ++i) {
            float rm = fmaxf(sfrag[0][i], sfrag[1][i]);
            #pragma unroll
            for (int m = 1; m < 16; m <<= 1) rm = fmaxf(rm, __shfl_xor(rm, m, 32));
            float mn    = fmaxf(mrow[i], rm);
            float alpha = __expf(mrow[i] - mn);
            p0[i] = __expf(sfrag[0][i] - mn);
            p1[i] = __expf(sfrag[1][i] - mn);
            float rs = p0[i] + p1[i];
            #pragma unroll
            for (int m = 1; m < 16; m <<= 1) rs += __shfl_xor(rs, m, 32);
            lrow[i] = lrow[i] * alpha + rs;
            mrow[i] = mn;
            #pragma unroll
            for (int n = 0; n < 4; ++n) o[n][i] *= alpha;
        }

        // P: C layout -> LDS [16][32] row-major -> A fragment (wave-private slice)
        #pragma unroll
        for (int i = 0; i < 8; ++i) {
            int m = i + 8 * half_id;
            myP[m * 32 + l16]      = (_Float16)p0[i];
            myP[m * 32 + 16 + l16] = (_Float16)p1[i];
        }
        asm volatile("s_wait_dscnt 0" ::: "memory");
        v16h pa;
        {
            const _Float16* pp = &myP[l16 * 32];
            v8h lo = *(const v8h*)(pp + koff);
            v8h hi = *(const v8h*)(pp + koff + 8);
            #pragma unroll
            for (int j = 0; j < 8; ++j) { pa[j] = lo[j]; pa[8 + j] = hi[j]; }
        }

        // O += P V  (V^T stored [B][D][T] -> contiguous B fragments)
        const _Float16* vp = vt + (size_t)b * D_ * T_ + s0 + half_id * 16;
        #pragma unroll
        for (int n = 0; n < 4; ++n) {
            int d = n * 16 + l16;
            v16h vb = *(const v16h*)(vp + (size_t)d * T_);
            o[n] = WMMA_F16(pa, vb, o[n]);
        }
    }

    // out = O / l
    #pragma unroll
    for (int n = 0; n < 4; ++n) {
        int d = n * 16 + l16;
        #pragma unroll
        for (int i = 0; i < 8; ++i) {
            int t = t0 + i + 8 * half_id;
            out[((size_t)b * T_ + t) * D_ + d] = o[n][i] / lrow[i];
        }
    }
}

// ---------------------------------------------------------------------------
extern "C" void kernel_launch(void* const* d_in, const int* in_sizes, int n_in,
                              void* d_out, int out_size, void* d_ws, size_t ws_size,
                              hipStream_t stream) {
    const float* x  = (const float*)d_in[0];
    const float* Wq = (const float*)d_in[1];
    const float* Wk = (const float*)d_in[2];
    const float* Wv = (const float*)d_in[3];
    float* out = (float*)d_out;

    // workspace layout (f16): wt[3*D*E] | qh[BT*D] | kh[BT*D] | vt[B*D*T]  (~6.7 MB)
    _Float16* wt = (_Float16*)d_ws;
    _Float16* qh = wt + (size_t)3 * D_ * E_;
    _Float16* kh = qh + (size_t)BT_ * D_;
    _Float16* vt = kh + (size_t)BT_ * D_;

    wconvert_kernel<<<(3 * D_ * E_ + 255) / 256, 256, 0, stream>>>(Wq, Wk, Wv, wt);
    qkv_proj_kernel<<<BT_ / (16 * 4), 128, 0, stream>>>(x, wt, qh, kh, vt);
    dim3 grid(T_ / (16 * 4), B_);
    attn_kernel<<<grid, 128, 0, stream>>>(qh, kh, vt, out);
}
